// EdgeDegreeEmbedding_84859963834920
// MI455X (gfx1250) — matrix operand order repaired and checked by reading, not compiled, not run on hardware
//
#include <hip/hip_runtime.h>
#include <hip/hip_bf16.h>

typedef __attribute__((ext_vector_type(16))) _Float16 v16h;
typedef __attribute__((ext_vector_type(8)))  float    v8f;
typedef __attribute__((ext_vector_type(4)))  float    f32x4;

#define HDIM     128
#define DIN      384
#define NOUT2    160
#define KFULL    25
#define NCOEF    19
#define RESC_INV 0.2f
#define MTILE    32     // edges per block (two 16-row WMMA M-tiles)

union HFrag { v16h v; f32x4 f4[2]; _Float16 h[16]; };
union FAcc  { v8f  v; float f[8]; };

// ---------------- pre-pass kernels ----------------

__global__ void zero_f32(float* p, int n) {
    int i = blockIdx.x * blockDim.x + threadIdx.x;
    if (i < n) p[i] = 0.0f;
}

// Convert W [K][N] f32 row-major -> f16 fragments laid out for the
// v_wmma_f32_16x16x32_f16 B operand in the contiguous k order the A loads use, per lane:
//   frag(tn,kk): lane L holds N = tn*16 + (L&15), K = kk*32 + (L<16?0:16) + j
//   (j = 0..15 contiguous) -> 16 contiguous halves (32B) per lane.
__global__ void swizzle_w(const float* __restrict__ src, _Float16* __restrict__ dst,
                          int K, int N) {
    int idx = blockIdx.x * blockDim.x + threadIdx.x;
    int total = K * N;
    if (idx >= total) return;
    int nK      = K >> 5;          // k-steps of 32
    int perTile = nK * 512;        // 32 lanes * 16 halves per k-step
    int tn   = idx / perTile;
    int rem  = idx - tn * perTile;
    int kk   = rem >> 9;
    int rem2 = rem & 511;
    int lane = rem2 >> 4;
    int j    = rem2 & 15;
    int n = tn * 16 + (lane & 15);
    int k = kk * 32 + ((lane < 16) ? 0 : 16) + j;
    dst[idx] = (_Float16)src[k * N + n];
}

// ---------------- device helpers ----------------

// Wave-cooperative GEMM: A is [32 x KDIM] f16 in LDS (row-major, stride KDIM),
// Wf is pre-swizzled B fragments, out is f32 LDS [32 x outStride].
// Wave w computes n-tiles tn = w, w+4, ... < NTILES; each B fragment is reused
// for both 16-row M-halves with independent accumulators (2 interleaved
// dependency chains keep the matrix pipe busy).
template <int KDIM, int NTILES>
__device__ __forceinline__
void wmma_gemm(const _Float16* __restrict__ A, const _Float16* __restrict__ Wf,
               const float* __restrict__ bias, float* __restrict__ outBuf,
               int outStride, float scale, int wave, int lane) {
    constexpr int NK = KDIM / 32;
    const int mA    = lane & 15;
    const int kbA   = (lane < 16) ? 0 : 16;  // A lane K-base (contiguous, matches B)
    const int nB    = lane & 15;
    const int mbase = (lane < 16) ? 0 : 8;   // D vgpr r -> row m = r + mbase

    for (int tn = wave; tn < NTILES; tn += 4) {
        FAcc acc0, acc1;
#pragma unroll
        for (int r = 0; r < 8; ++r) { acc0.f[r] = 0.0f; acc1.f[r] = 0.0f; }
#pragma unroll
        for (int kk = 0; kk < NK; ++kk) {
            HFrag a0, a1, b;
            const _Float16* ap0 = A + mA * KDIM + kk * 32 + kbA;
            const _Float16* ap1 = ap0 + 16 * KDIM;          // rows 16..31
            a0.f4[0] = *(const f32x4*)(ap0);
            a0.f4[1] = *(const f32x4*)(ap0 + 8);
            a1.f4[0] = *(const f32x4*)(ap1);
            a1.f4[1] = *(const f32x4*)(ap1 + 8);
            const _Float16* bp = Wf + (((size_t)tn * NK + kk) * 32 + lane) * 16;
            b.f4[0] = *(const f32x4*)(bp);
            b.f4[1] = *(const f32x4*)(bp + 8);
            acc0.v = __builtin_amdgcn_wmma_f32_16x16x32_f16(
                false, a0.v, false, b.v, (short)0, acc0.v, false, false);
            acc1.v = __builtin_amdgcn_wmma_f32_16x16x32_f16(
                false, a1.v, false, b.v, (short)0, acc1.v, false, false);
        }
        int n = tn * 16 + nB;
        float bv = bias[n];
#pragma unroll
        for (int r = 0; r < 8; ++r) {
            outBuf[(mbase + r) * outStride + n]      = (acc0.f[r] + bv) * scale;
            outBuf[(16 + mbase + r) * outStride + n] = (acc1.f[r] + bv) * scale;
        }
    }
}

// Per-row LayerNorm + SiLU; wave w handles rows 8w..8w+7, 4 cols per lane.
__device__ __forceinline__
void ln_silu_store(const float* __restrict__ src, _Float16* __restrict__ dst,
                   const float* __restrict__ g, const float* __restrict__ bb,
                   int wave, int lane) {
    f32x4 gv = *(const f32x4*)(g + lane * 4);
    f32x4 bv = *(const f32x4*)(bb + lane * 4);
#pragma unroll
    for (int q = 0; q < 8; ++q) {
        int row = wave * 8 + q;
        f32x4 v = *(const f32x4*)(src + row * HDIM + lane * 4);
        float s  = v.x + v.y + v.z + v.w;
        float s2 = v.x * v.x + v.y * v.y + v.z * v.z + v.w * v.w;
#pragma unroll
        for (int off = 16; off > 0; off >>= 1) {
            s  += __shfl_xor(s,  off);
            s2 += __shfl_xor(s2, off);
        }
        float mu   = s  * (1.0f / HDIM);
        float var  = s2 * (1.0f / HDIM) - mu * mu;
        float rstd = rsqrtf(var + 1e-6f);
        float y0 = (v.x - mu) * rstd * gv.x + bv.x;
        float y1 = (v.y - mu) * rstd * gv.y + bv.y;
        float y2 = (v.z - mu) * rstd * gv.z + bv.z;
        float y3 = (v.w - mu) * rstd * gv.w + bv.w;
        y0 = y0 / (1.0f + __expf(-y0));
        y1 = y1 / (1.0f + __expf(-y1));
        y2 = y2 / (1.0f + __expf(-y2));
        y3 = y3 / (1.0f + __expf(-y3));
        _Float16* dp = dst + row * HDIM + lane * 4;
        dp[0] = (_Float16)y0; dp[1] = (_Float16)y1;
        dp[2] = (_Float16)y2; dp[3] = (_Float16)y3;
    }
}

// ---------------- main edge kernel: 32 edges per block, 4 waves ----------------

__global__ __launch_bounds__(128)
void edge_kernel(const float* __restrict__ edge_embeds,
                 const int*   __restrict__ senders,
                 const int*   __restrict__ receivers,
                 const float* __restrict__ wigner_inv,
                 const float* __restrict__ embed_table,
                 const int*   __restrict__ node_species,
                 const _Float16* __restrict__ W0h, const float* __restrict__ b0,
                 const float* __restrict__ ln0_s, const float* __restrict__ ln0_b,
                 const _Float16* __restrict__ W1h, const float* __restrict__ b1,
                 const float* __restrict__ ln1_s, const float* __restrict__ ln1_b,
                 const _Float16* __restrict__ W2h, const float* __restrict__ b2,
                 float* __restrict__ out, int E) {
    __shared__ __align__(16) _Float16 xin[MTILE * DIN];     // 24 KB f16 input tile
    __shared__ __align__(16) float    hbuf[MTILE * HDIM];   // 16 KB f32 GEMM out
    __shared__ __align__(16) _Float16 x1[MTILE * HDIM];     //  8 KB
    __shared__ __align__(16) _Float16 x2[MTILE * HDIM];     //  8 KB
    __shared__ __align__(16) float    feats[MTILE * NOUT2]; // 20 KB
    __shared__ int sspec[MTILE], rspec[MTILE], rnode[MTILE];

    const int tid  = threadIdx.x;
    const int wave = tid >> 5;
    const int lane = tid & 31;
    const int e0   = blockIdx.x * MTILE;

    // Prefetch this block's wigner_inv rows (the dominant HBM stream, consumed
    // only after 3 GEMMs + 2 LNs) so stage E hits cache instead of HBM.
    {
        int nve = E - e0; if (nve > MTILE) nve = MTILE;
        if (nve > 0) {
            const char* wbase = (const char*)(wigner_inv + (size_t)e0 * (KFULL * NCOEF));
            int wbytes = nve * (KFULL * NCOEF) * 4;
            for (int off = tid * 128; off < wbytes; off += 128 * 128)
                __builtin_prefetch(wbase + off, 0, 1);
        }
    }

    if (tid < MTILE) {
        int ee = e0 + tid; if (ee > E - 1) ee = E - 1;
        sspec[tid] = node_species[senders[ee]];
        rspec[tid] = node_species[receivers[ee]];
        rnode[tid] = receivers[ee];
    }
    __syncthreads();

    // Build x = concat(edge_embed, s_emb, r_emb) as f16; 3072 float4 loads.
    for (int idx = tid; idx < MTILE * 96; idx += 128) {
        int e = idx / 96;
        int q = idx - e * 96;
        int c = q * 4;
        int ee = e0 + e; if (ee > E - 1) ee = E - 1;
        f32x4 v;
        if (c < 128)
            v = *(const f32x4*)(edge_embeds + (size_t)ee * 128 + c);
        else if (c < 256)
            v = *(const f32x4*)(embed_table + (size_t)sspec[e] * 256 + (c - 128));
        else
            v = *(const f32x4*)(embed_table + (size_t)rspec[e] * 256 + 128 + (c - 256));
        _Float16* xp = xin + e * DIN + c;
        xp[0] = (_Float16)v.x; xp[1] = (_Float16)v.y;
        xp[2] = (_Float16)v.z; xp[3] = (_Float16)v.w;
    }
    __syncthreads();

    // GEMM0: [32,384] @ [384,128]  -> hbuf
    wmma_gemm<DIN, 8>(xin, W0h, b0, hbuf, HDIM, 1.0f, wave, lane);
    __syncthreads();
    ln_silu_store(hbuf, x1, ln0_s, ln0_b, wave, lane);
    __syncthreads();

    // GEMM1: [32,128] @ [128,128] -> hbuf
    wmma_gemm<HDIM, 8>(x1, W1h, b1, hbuf, HDIM, 1.0f, wave, lane);
    __syncthreads();
    ln_silu_store(hbuf, x2, ln1_s, ln1_b, wave, lane);
    __syncthreads();

    // GEMM2: [32,128] @ [128,160] -> feats, fused bias + 1/RESCALE
    wmma_gemm<HDIM, 10>(x2, W2h, b2, feats, NOUT2, RESC_INV, wave, lane);
    __syncthreads();

    // Wigner rotation (only m=0..4 columns are non-zero) + segment-sum scatter.
    for (int flat = tid; flat < MTILE * KFULL; flat += 128) {
        int e = flat / KFULL;
        int k = flat - e * KFULL;
        if (e0 + e >= E) continue;
        const float* wp = wigner_inv + ((size_t)(e0 + e) * KFULL + k) * NCOEF;
        float w0 = wp[0], w1 = wp[1], w2 = wp[2], w3 = wp[3], w4 = wp[4];
        const float* fp = feats + e * NOUT2;
        float* op = out + (size_t)rnode[e] * (KFULL * 32) + k * 32;
#pragma unroll 4
        for (int c = 0; c < 32; ++c) {
            float sv = w0 * fp[c] + w1 * fp[32 + c] + w2 * fp[64 + c] +
                       w3 * fp[96 + c] + w4 * fp[128 + c];
            atomicAdd(op + c, sv);
        }
    }
}

// ---------------- host-side launch ----------------

extern "C" void kernel_launch(void* const* d_in, const int* in_sizes, int n_in,
                              void* d_out, int out_size, void* d_ws, size_t ws_size,
                              hipStream_t stream) {
    const int*   node_species = (const int*)  d_in[0];
    const float* edge_embeds  = (const float*)d_in[1];
    const int*   senders      = (const int*)  d_in[2];
    const int*   receivers    = (const int*)  d_in[3];
    const float* wigner_inv   = (const float*)d_in[4];
    const float* embed_table  = (const float*)d_in[5];
    const float* W0    = (const float*)d_in[6];
    const float* b0    = (const float*)d_in[7];
    const float* ln0_s = (const float*)d_in[8];
    const float* ln0_b = (const float*)d_in[9];
    const float* W1    = (const float*)d_in[10];
    const float* b1    = (const float*)d_in[11];
    const float* ln1_s = (const float*)d_in[12];
    const float* ln1_b = (const float*)d_in[13];
    const float* W2    = (const float*)d_in[14];
    const float* b2    = (const float*)d_in[15];
    float* out = (float*)d_out;

    const int E = in_sizes[1] / 128;   // edge_embeds is [E,128]

    // f16 pre-swizzled weights in workspace
    _Float16* W0h = (_Float16*)d_ws;                          // 384*128 halves
    _Float16* W1h = (_Float16*)((char*)d_ws + 98304);         // 128*128 halves
    _Float16* W2h = (_Float16*)((char*)d_ws + 131072);        // 128*160 halves

    zero_f32<<<(out_size + 255) / 256, 256, 0, stream>>>(out, out_size);
    swizzle_w<<<(384 * 128 + 255) / 256, 256, 0, stream>>>(W0, W0h, 384, 128);
    swizzle_w<<<(128 * 128 + 255) / 256, 256, 0, stream>>>(W1, W1h, 128, 128);
    swizzle_w<<<(128 * 160 + 255) / 256, 256, 0, stream>>>(W2, W2h, 128, 160);

    int nTiles = (E + MTILE - 1) / MTILE;
    edge_kernel<<<nTiles, 128, 0, stream>>>(
        edge_embeds, senders, receivers, wigner_inv, embed_table, node_species,
        W0h, b0, ln0_s, ln0_b, W1h, b1, ln1_s, ln1_b, W2h, b2, out, E);
}
